// NodeBlockLite_86844238725708
// MI455X (gfx1250) — matrix operand order repaired and checked by reading, not compiled, not run on hardware
//
#include <hip/hip_runtime.h>
#include <hip/hip_bf16.h>
#include <math.h>

// ---------------- problem constants ----------------
#define B_   16
#define N_   4096
#define ROWS 64              // node rows per workgroup

typedef __attribute__((ext_vector_type(16))) __bf16 v16bf;
typedef __attribute__((ext_vector_type(8)))  float  v8f;

__device__ __forceinline__ unsigned short f2bf(float f) {
  unsigned u = __float_as_uint(f);
  return (unsigned short)((u + 0x7fffu + ((u >> 16) & 1u)) >> 16);  // RNE
}
__device__ __forceinline__ float lrelu(float x) { return x > 0.f ? x : 0.01f * x; }
#define NEG_INF __int_as_float(0xff800000)

// ---------------- workspace layout (bytes) ----------------
// W1S: feat_w1 bf16 B-frags  [384x256]  196608
// W2S: feat_w2 bf16 B-frags  [256x128]   65536
// A1S: attn_w1 bf16 B-frags  [384x128]   98304
// A2S: attn_w2 bf16 B-frags  [128x16 ]    4096  (cols 4..15 zero padded)
// HL : [B,128] f32 concat(globs,ctxt)     8192
// LOG: [B,N,4] f32 masked logits       1048576
#define WS_W1S 0
#define WS_W2S 196608
#define WS_A1S 262144
#define WS_A2S 360448
#define WS_HL  364544
#define WS_LOG 372736

// B-fragment blob: 1024 B = 32 lanes x 16 bf16.  lane = (n&15) + 16*((k>>3)&1),
// half j = (k&7) + 8*((k>>4)&1)  ->  each lane's 16 halfs are contiguous (two b128 loads).
__global__ void swizzle_w(const float* __restrict__ src, unsigned short* __restrict__ dst,
                          int K, int srcN, int dstN) {
  int id = blockIdx.x * 256 + threadIdx.x;
  if (id >= K * dstN) return;
  int k = id / dstN, n = id % dstN;
  float v = (n < srcN) ? src[k * srcN + n] : 0.f;
  int NT = dstN >> 4;
  int blob = (k >> 5) * NT + (n >> 4);
  int lanew = (n & 15) + (((k >> 3) & 1) << 4);
  dst[(blob << 9) + (lanew << 4) + ((k & 7) + 8 * ((k >> 4) & 1))] = f2bf(v);
}

__global__ void build_hl(const float* __restrict__ g, const float* __restrict__ c,
                         float* __restrict__ hl) {
  int id = blockIdx.x * 256 + threadIdx.x;
  if (id >= B_ * 128) return;
  int b = id >> 7, cc = id & 127;
  hl[id] = (cc < 64) ? g[b * 64 + cc] : c[b * 64 + (cc - 64)];
}

// ---------------- main fused kernel ----------------
// LDS (dynamic, 98304 B):
//   XIN: 4 mt x 12 kt blobs (bf16 A-frags of xin[64,384])   48 KB
//   H1 : 4 mt x  8 kt blobs (hidden feat [64,256])           32 KB
//   A1 : 4 mt x  4 kt blobs (hidden attn [64,128])           16 KB
__global__ __launch_bounds__(256) void nodeblock_main(
    const float* __restrict__ nodes, const float* __restrict__ pedges,
    const unsigned char* __restrict__ maskp,
    const float* __restrict__ lng, const float* __restrict__ lnb,
    const float* __restrict__ fb1, const float* __restrict__ fb2,
    const float* __restrict__ ab1, const float* __restrict__ ab2,
    const unsigned short* __restrict__ w1s, const unsigned short* __restrict__ w2s,
    const unsigned short* __restrict__ a1s, const unsigned short* __restrict__ a2s,
    const float* __restrict__ hl,
    float* __restrict__ out_nodes, float* __restrict__ logits) {
  extern __shared__ char smem[];
  char* XIN = smem;
  char* H1  = smem + 49152;
  char* A1  = smem + 81920;

  const int tid = threadIdx.x, wave = tid >> 5, lane = tid & 31;
  const int g0 = blockIdx.x * ROWS;
  const int batch = g0 >> 12;  // N_ = 4096

  // ===== stage 1: LayerNorm + xin build (A-frag bf16 layout) =====
  const int c0 = lane * 8;  // lanes 0-15 -> nodes cols, 16-31 -> pooled_edges cols
  float4 ga = *(const float4*)(lng + c0), gb = *(const float4*)(lng + c0 + 4);
  float4 ba = *(const float4*)(lnb + c0), bb = *(const float4*)(lnb + c0 + 4);

  // hl slice: cols 256 + lane*4 .. +3 (same for every row of this batch)
  float4 hv = *(const float4*)(hl + batch * 128 + lane * 4);
  unsigned hl_lo = (unsigned)f2bf(hv.x) | ((unsigned)f2bf(hv.y) << 16);
  unsigned hl_hi = (unsigned)f2bf(hv.z) | ((unsigned)f2bf(hv.w) << 16);
  const int hcol = 256 + lane * 4, hkt = hcol >> 5, hkk = hcol & 31;
  const int hkb = (hkk >> 3) & 1, hj = (hkk & 7) + 8 * ((hkk >> 4) & 1);

  const int kt1 = c0 >> 5, kk1 = c0 & 31;
  const int kb1 = (kk1 >> 3) & 1, jb1 = 8 * ((kk1 >> 4) & 1);

  for (int rr = 0; rr < 8; ++rr) {
    const int row = wave * 8 + rr, gnode = g0 + row;
    const int mt = row >> 4, m = row & 15;
    const float* src = (lane < 16) ? (nodes + (size_t)gnode * 128 + c0)
                                   : (pedges + (size_t)gnode * 128 + (c0 - 128));
    float4 v0 = *(const float4*)(src);
    float4 v1 = *(const float4*)(src + 4);
    float s = v0.x + v0.y + v0.z + v0.w + v1.x + v1.y + v1.z + v1.w;
    float q = v0.x*v0.x + v0.y*v0.y + v0.z*v0.z + v0.w*v0.w
            + v1.x*v1.x + v1.y*v1.y + v1.z*v1.z + v1.w*v1.w;
    for (int off = 16; off; off >>= 1) {
      s += __shfl_xor(s, off, 32);
      q += __shfl_xor(q, off, 32);
    }
    float mean = s * (1.f / 256.f);
    float rstd = rsqrtf(q * (1.f / 256.f) - mean * mean + 1e-5f);
    float y0 = (v0.x - mean) * rstd * ga.x + ba.x;
    float y1 = (v0.y - mean) * rstd * ga.y + ba.y;
    float y2 = (v0.z - mean) * rstd * ga.z + ba.z;
    float y3 = (v0.w - mean) * rstd * ga.w + ba.w;
    float y4 = (v1.x - mean) * rstd * gb.x + bb.x;
    float y5 = (v1.y - mean) * rstd * gb.y + bb.y;
    float y6 = (v1.z - mean) * rstd * gb.z + bb.z;
    float y7 = (v1.w - mean) * rstd * gb.w + bb.w;
    uint4 pk;
    pk.x = (unsigned)f2bf(y0) | ((unsigned)f2bf(y1) << 16);
    pk.y = (unsigned)f2bf(y2) | ((unsigned)f2bf(y3) << 16);
    pk.z = (unsigned)f2bf(y4) | ((unsigned)f2bf(y5) << 16);
    pk.w = (unsigned)f2bf(y6) | ((unsigned)f2bf(y7) << 16);
    *(uint4*)(XIN + ((mt * 12 + kt1) << 10) + ((m + 16 * kb1) << 5) + jb1 * 2) = pk;
    *(uint2*)(XIN + ((mt * 12 + hkt) << 10) + ((m + 16 * hkb) << 5) + hj * 2) =
        make_uint2(hl_lo, hl_hi);
  }
  __syncthreads();

  // ===== stage 2: fused GEMM1 (feat 384->256) + GEMM1 (attn 384->128) =====
  const int mt = wave >> 1;
  const int nt0f = (wave & 1) * 8;   // feat hidden: 8 n-tiles of 16
  const int nt0h = (wave & 1) * 4;   // attn hidden / 128-wide outs: 4 n-tiles
  v8f zf = {0.f, 0.f, 0.f, 0.f, 0.f, 0.f, 0.f, 0.f};
  v8f accF[8], accA[4];
#pragma unroll
  for (int t = 0; t < 8; ++t) accF[t] = zf;
#pragma unroll
  for (int t = 0; t < 4; ++t) accA[t] = zf;

  for (int kt = 0; kt < 12; ++kt) {
    v16bf a = *(const v16bf*)(XIN + ((mt * 12 + kt) << 10) + (lane << 5));
#pragma unroll
    for (int t = 0; t < 8; ++t) {
      v16bf bw = *(const v16bf*)(w1s + (((kt << 4) + nt0f + t) << 9) + (lane << 4));
      accF[t] = __builtin_amdgcn_wmma_f32_16x16x32_bf16(false, a, false, bw,
                                                        (short)0, accF[t], false, false);
    }
#pragma unroll
    for (int t = 0; t < 4; ++t) {
      v16bf bw = *(const v16bf*)(a1s + (((kt << 3) + nt0h + t) << 9) + (lane << 4));
      accA[t] = __builtin_amdgcn_wmma_f32_16x16x32_bf16(false, a, false, bw,
                                                        (short)0, accA[t], false, false);
    }
  }

  // epilogue: +bias, LeakyReLU, redeposit as bf16 A-frags
  const int nn = lane & 15, kbl = lane >> 4;
#pragma unroll
  for (int t = 0; t < 8; ++t) {
    int ncol = (nt0f + t) * 16 + nn;
    float bias = fb1[ncol];
    int kt2 = ncol >> 5, kk = ncol & 31;
    int kb2 = (kk >> 3) & 1, j = (kk & 7) + 8 * ((kk >> 4) & 1);
    char* dst = H1 + ((mt * 8 + kt2) << 10) + j * 2;
#pragma unroll
    for (int r = 0; r < 8; ++r) {
      int row16 = r + 8 * kbl;
      *(unsigned short*)(dst + ((row16 + 16 * kb2) << 5)) = f2bf(lrelu(accF[t][r] + bias));
    }
  }
#pragma unroll
  for (int t = 0; t < 4; ++t) {
    int ncol = (nt0h + t) * 16 + nn;
    float bias = ab1[ncol];
    int kt2 = ncol >> 5, kk = ncol & 31;
    int kb2 = (kk >> 3) & 1, j = (kk & 7) + 8 * ((kk >> 4) & 1);
    char* dst = A1 + ((mt * 4 + kt2) << 10) + j * 2;
#pragma unroll
    for (int r = 0; r < 8; ++r) {
      int row16 = r + 8 * kbl;
      *(unsigned short*)(dst + ((row16 + 16 * kb2) << 5)) = f2bf(lrelu(accA[t][r] + bias));
    }
  }
  __syncthreads();

  // ===== stage 3: GEMM2 feat (256->128) + residual; attn head (128->16pad) =====
  v8f accF2[4];
#pragma unroll
  for (int t = 0; t < 4; ++t) accF2[t] = zf;
  for (int kt = 0; kt < 8; ++kt) {
    v16bf a = *(const v16bf*)(H1 + ((mt * 8 + kt) << 10) + (lane << 5));
#pragma unroll
    for (int t = 0; t < 4; ++t) {
      v16bf bw = *(const v16bf*)(w2s + (((kt << 3) + nt0h + t) << 9) + (lane << 4));
      accF2[t] = __builtin_amdgcn_wmma_f32_16x16x32_bf16(false, a, false, bw,
                                                         (short)0, accF2[t], false, false);
    }
  }
#pragma unroll
  for (int t = 0; t < 4; ++t) {
    int ncol = (nt0h + t) * 16 + nn;
    float bias = fb2[ncol];
#pragma unroll
    for (int r = 0; r < 8; ++r) {
      int row = mt * 16 + r + 8 * kbl;
      int gnode = g0 + row;
      float f = accF2[t][r] + bias;
      f = maskp[gnode] ? f : 0.f;                     // zero-pad feats
      size_t idx = (size_t)gnode * 128 + ncol;
      out_nodes[idx] = f + nodes[idx];                // residual
    }
  }

  // attention logits: [64,128] x [128,16(pad)] ; only cols 0..3 valid
  if (wave < 4) {
    const int mta = wave;
    v8f accL = zf;
    for (int kt = 0; kt < 4; ++kt) {
      v16bf a  = *(const v16bf*)(A1 + ((mta * 4 + kt) << 10) + (lane << 5));
      v16bf bw = *(const v16bf*)(a2s + ((size_t)kt << 9) + (lane << 4));
      accL = __builtin_amdgcn_wmma_f32_16x16x32_bf16(false, a, false, bw,
                                                     (short)0, accL, false, false);
    }
    if (nn < 4) {
      float bias = ab2[nn];
#pragma unroll
      for (int r = 0; r < 8; ++r) {
        int row = mta * 16 + r + 8 * kbl;
        int gnode = g0 + row;
        float v = accL[r] + bias;
        logits[(size_t)gnode * 4 + nn] = maskp[gnode] ? v : NEG_INF;
      }
    }
  }
}

// ---------------- softmax over node axis + weighted pooling ----------------
__global__ __launch_bounds__(256) void softmax_pool(
    const float* __restrict__ logits, const float* __restrict__ nnodes,
    float* __restrict__ pooled) {
  __shared__ float sred[8][4];
  __shared__ float sacc[256];
  const int b = blockIdx.x;
  const int tid = threadIdx.x, wave = tid >> 5, lane = tid & 31;
  const float* lg = logits + (size_t)b * N_ * 4;

  // pass 1: per-head max (invalid nodes are -inf)
  float mx[4] = {NEG_INF, NEG_INF, NEG_INF, NEG_INF};
  for (int n = tid; n < N_; n += 256) {
    const float* lp = lg + (size_t)n * 4;
#pragma unroll
    for (int h = 0; h < 4; ++h) mx[h] = fmaxf(mx[h], lp[h]);
  }
#pragma unroll
  for (int h = 0; h < 4; ++h)
    for (int off = 16; off; off >>= 1) mx[h] = fmaxf(mx[h], __shfl_xor(mx[h], off, 32));
  if (lane == 0) {
#pragma unroll
    for (int h = 0; h < 4; ++h) sred[wave][h] = mx[h];
  }
  __syncthreads();
#pragma unroll
  for (int h = 0; h < 4; ++h) {
    float m = sred[0][h];
    for (int w = 1; w < 8; ++w) m = fmaxf(m, sred[w][h]);
    mx[h] = m;
  }
  __syncthreads();

  // pass 2: sum of exp
  float sm[4] = {0.f, 0.f, 0.f, 0.f};
  for (int n = tid; n < N_; n += 256) {
    const float* lp = lg + (size_t)n * 4;
#pragma unroll
    for (int h = 0; h < 4; ++h) sm[h] += __expf(lp[h] - mx[h]);
  }
#pragma unroll
  for (int h = 0; h < 4; ++h)
    for (int off = 16; off; off >>= 1) sm[h] += __shfl_xor(sm[h], off, 32);
  if (lane == 0) {
#pragma unroll
    for (int h = 0; h < 4; ++h) sred[wave][h] = sm[h];
  }
  __syncthreads();
  float scale[4];
#pragma unroll
  for (int h = 0; h < 4; ++h) {
    float t = 0.f;
    for (int w = 0; w < 8; ++w) t += sred[w][h];
    scale[h] = 1.f / (t * 11.313708499f);  // softmax / sqrt(128)
  }

  // pass 3: pooled[c] = sum_n new_nodes[n][c] * w[n][c/32]
  int c = tid & 127, half = tid >> 7, h = c >> 5;
  float acc = 0.f;
  const float* nb = nnodes + (size_t)b * N_ * 128;
  for (int n = half; n < N_; n += 2) {
    float wgt = __expf(lg[(size_t)n * 4 + h] - mx[h]) * scale[h];
    acc += nb[(size_t)n * 128 + c] * wgt;
  }
  sacc[tid] = acc;
  __syncthreads();
  if (tid < 128) pooled[(size_t)b * 128 + tid] = sacc[tid] + sacc[tid + 128];
}

// ---------------- launcher ----------------
extern "C" void kernel_launch(void* const* d_in, const int* in_sizes, int n_in,
                              void* d_out, int out_size, void* d_ws, size_t ws_size,
                              hipStream_t stream) {
  const float* nodes  = (const float*)d_in[0];
  const float* pedges = (const float*)d_in[1];
  const unsigned char* mask = (const unsigned char*)d_in[2];  // jax bool
  const float* globs = (const float*)d_in[3];
  const float* ctxt  = (const float*)d_in[4];
  const float* lng   = (const float*)d_in[5];
  const float* lnb   = (const float*)d_in[6];
  const float* fw1   = (const float*)d_in[7];
  const float* fb1   = (const float*)d_in[8];
  const float* fw2   = (const float*)d_in[9];
  const float* fb2   = (const float*)d_in[10];
  const float* aw1   = (const float*)d_in[11];
  const float* ab1   = (const float*)d_in[12];
  const float* aw2   = (const float*)d_in[13];
  const float* ab2   = (const float*)d_in[14];

  char* ws = (char*)d_ws;
  unsigned short* w1s = (unsigned short*)(ws + WS_W1S);
  unsigned short* w2s = (unsigned short*)(ws + WS_W2S);
  unsigned short* a1s = (unsigned short*)(ws + WS_A1S);
  unsigned short* a2s = (unsigned short*)(ws + WS_A2S);
  float* hl     = (float*)(ws + WS_HL);
  float* logits = (float*)(ws + WS_LOG);

  float* out_nodes = (float*)d_out;
  float* pooled = out_nodes + (size_t)B_ * N_ * 128;

  build_hl<<<8, 256, 0, stream>>>(globs, ctxt, hl);
  swizzle_w<<<(384 * 256 + 255) / 256, 256, 0, stream>>>(fw1, w1s, 384, 256, 256);
  swizzle_w<<<(256 * 128 + 255) / 256, 256, 0, stream>>>(fw2, w2s, 256, 128, 128);
  swizzle_w<<<(384 * 128 + 255) / 256, 256, 0, stream>>>(aw1, a1s, 384, 128, 128);
  swizzle_w<<<(128 * 16 + 255) / 256, 256, 0, stream>>>(aw2, a2s, 128, 4, 16);

  nodeblock_main<<<(B_ * N_) / ROWS, 256, 98304, stream>>>(
      nodes, pedges, mask, lng, lnb, fb1, fb2, ab1, ab2,
      w1s, w2s, a1s, a2s, hl, out_nodes, logits);

  softmax_pool<<<B_, 256, 0, stream>>>(logits, out_nodes, pooled);
}
